// SiameseEdgeConvNet_30880814859093
// MI455X (gfx1250) — matrix operand order repaired and checked by reading, not compiled, not run on hardware
//
#include <hip/hip_runtime.h>
#include <hip/hip_bf16.h>
#include <stdint.h>

// -----------------------------------------------------------------------------
// Siamese EdgeConv on MI455X (gfx1250, wave32)
//   - per-edge MLP tensorized with v_wmma_f32_16x16x32_bf16 (16 edges/wave)
//   - segment_max via native no-return global_atomic_max_num_f32 into a -inf
//     seeded buffer (IEEE maximumNumber semantics; messages are relu'd >= 0)
//   - finalize pass: where(isfinite, v, 0) (+ PReLU after layer 1)
// Compute (~79 GFLOP) is trivial; node tensors are L2-resident (192MB L2),
// so the kernel is L2-gather/atomic bound. WMMA removes the ALU cost of the
// per-edge GEMMs; bf16 in / f32 accumulate.
// -----------------------------------------------------------------------------

#define GN 50000
#define GE 1600000

typedef __attribute__((ext_vector_type(16))) __bf16 v16bf;
typedef __attribute__((ext_vector_type(8)))  float  v8f;
typedef __attribute__((ext_vector_type(4)))  float  f32x4;

__device__ __forceinline__ void atomic_max_f32_dev(float* p, float v) {
  asm volatile("global_atomic_max_num_f32 %0, %1, off scope:SCOPE_DEV"
               :: "v"((unsigned long long)(uintptr_t)p), "v"(v)
               : "memory");
}

__global__ void fill_kernel(float* __restrict__ p, int n, float val) {
  int i = blockIdx.x * blockDim.x + threadIdx.x;
  if (i < n) p[i] = val;
}

// h = isfinite(agg) ? prelu(agg, a[channel]) : 0
__global__ void finalize_prelu_kernel(const float* __restrict__ agg,
                                      const float* __restrict__ a,
                                      float* __restrict__ h, int n) {
  int i = blockIdx.x * blockDim.x + threadIdx.x;
  if (i < n) {
    float v = agg[i];
    float r = 0.0f;
    if (__builtin_isfinite(v)) r = (v >= 0.0f) ? v : a[i & 63] * v;
    h[i] = r;
  }
}

// in-place: out = isfinite(out) ? out : 0
__global__ void finalize_zero_kernel(float* __restrict__ agg, int n) {
  int i = blockIdx.x * blockDim.x + threadIdx.x;
  if (i < n) {
    float v = agg[i];
    agg[i] = __builtin_isfinite(v) ? v : 0.0f;
  }
}

// One wave handles 16 edges x 64 outputs. D = node feature dim (32 or 64);
// concat feature dim = 2*D; K-chunks of 32 for wmma_f32_16x16x32_bf16.
// ei32: edge_index int64 buffer viewed as int32 pairs (low word valid,
//       indices < 50000): src row = ei32[2*e], dst row = ei32[2*(E+e)].
template <int D>
__global__ __launch_bounds__(256)
void edgeconv_wmma_kernel(const float* __restrict__ X,
                          const int*   __restrict__ ei32,
                          const float* __restrict__ W,     // [2*D][64] row-major
                          const float* __restrict__ bias,  // [64]
                          float*       __restrict__ agg,   // [N*64], -inf seeded
                          int nEdges) {
  constexpr int KC = (2 * D) / 32;     // 2 (layer1) or 4 (layer2)
  const int lane = threadIdx.x & 31;
  const int ln15 = lane & 15;
  const bool hi  = lane >= 16;
  const int koff = hi ? 8 : 0;         // A-matrix per-lane K run start (ISA 7.12.2)

  // ---- B matrix (weights) resident in VGPRs, bf16, loaded once per wave ----
  // Lane = output column (N), K rows per lane match A's layout:
  //   lanes 0-15 -> K = chunk*32 + 0..7,16..23 ; lanes 16-31 -> K = chunk*32 + 8..15,24..31
  v16bf Bv[KC][4];
#pragma unroll
  for (int k = 0; k < KC; ++k) {
#pragma unroll
    for (int t = 0; t < 4; ++t) {
      const int col  = t * 16 + ln15;
      const int krow = k * 32 + (hi ? 8 : 0);
      v16bf b;
#pragma unroll
      for (int j = 0; j < 16; ++j)
        b[j] = (__bf16)W[(krow + j + (j >= 8 ? 8 : 0)) * 64 + col];
      Bv[k][t] = b;
    }
  }
  const float bcol[4] = { bias[ln15], bias[16 + ln15], bias[32 + ln15], bias[48 + ln15] };

  const int nTiles = (nEdges + 15) >> 4;
  const int waveId = blockIdx.x * (blockDim.x >> 5) + (threadIdx.x >> 5);
  const int nWaves = gridDim.x * (blockDim.x >> 5);

  const v8f vzero = {0.f, 0.f, 0.f, 0.f, 0.f, 0.f, 0.f, 0.f};

  for (int tile = waveId; tile < nTiles; tile += nWaves) {
    int e = tile * 16 + ln15;
    if (e >= nEdges) e = nEdges - 1;  // duplicate edge: atomic max is idempotent
    const int s = ei32[(size_t)2 * e];
    const int d = ei32[(size_t)2 * ((size_t)nEdges + e)];

    // prefetch next tile's edge indices (global_prefetch_b8)
    const int pe = (tile + nWaves) * 16 + ln15;
    if (pe < nEdges) {
      __builtin_prefetch(&ei32[(size_t)2 * pe], 0, 0);
      __builtin_prefetch(&ei32[(size_t)2 * ((size_t)nEdges + pe)], 0, 0);
    }

    const float* Xd = X + (size_t)d * D;
    const float* Xs = X + (size_t)s * D;

    v8f acc[4] = {vzero, vzero, vzero, vzero};

#pragma unroll
    for (int k = 0; k < KC; ++k) {
      const bool isDiff = (k >= KC / 2);           // chunks: [xi | xj - xi]
      const int  fb = (k & (KC / 2 - 1)) * 32;     // feature base within node row
      const int  p0 = fb + koff;                   // two contiguous 8-float runs
      const int  p1 = p0 + 16;

      f32x4 d0 = *(const f32x4*)(Xd + p0);
      f32x4 d1 = *(const f32x4*)(Xd + p0 + 4);
      f32x4 d2 = *(const f32x4*)(Xd + p1);
      f32x4 d3 = *(const f32x4*)(Xd + p1 + 4);
      if (isDiff) {
        d0 = *(const f32x4*)(Xs + p0)     - d0;
        d1 = *(const f32x4*)(Xs + p0 + 4) - d1;
        d2 = *(const f32x4*)(Xs + p1)     - d2;
        d3 = *(const f32x4*)(Xs + p1 + 4) - d3;
      }
      v16bf a;
#pragma unroll
      for (int j = 0; j < 4; ++j) {
        a[j]      = (__bf16)d0[j];
        a[4 + j]  = (__bf16)d1[j];
        a[8 + j]  = (__bf16)d2[j];
        a[12 + j] = (__bf16)d3[j];
      }
#pragma unroll
      for (int t = 0; t < 4; ++t)
        acc[t] = __builtin_amdgcn_wmma_f32_16x16x32_bf16(
            false, a, false, Bv[k][t], (short)0, acc[t], false, false);
    }

    // C layout: VGPR r -> row r (lanes 0-15) / row r+8 (lanes 16-31), lane = col.
#pragma unroll
    for (int t = 0; t < 4; ++t) {
      const float bb = bcol[t];
#pragma unroll
      for (int r = 0; r < 8; ++r) {
        float m = acc[t][r] + bb;
        m = m > 0.0f ? m : 0.0f;                 // relu
        const int row = r + (hi ? 8 : 0);
        const int dn  = __shfl(d, row, 32);      // dst node of that edge row
        atomic_max_f32_dev(agg + (size_t)dn * 64 + t * 16 + ln15, m);
      }
    }
  }
}

extern "C" void kernel_launch(void* const* d_in, const int* in_sizes, int n_in,
                              void* d_out, int out_size, void* d_ws, size_t ws_size,
                              hipStream_t stream) {
  const float* x1  = (const float*)d_in[0];
  const int*   ei1 = (const int*)d_in[1];   // int64 viewed as int32 pairs
  const float* x2  = (const float*)d_in[2];
  const int*   ei2 = (const int*)d_in[3];
  const float* W1  = (const float*)d_in[4]; // [64][64]
  const float* b1  = (const float*)d_in[5];
  const float* pa  = (const float*)d_in[6]; // prelu_a [64]
  const float* W2  = (const float*)d_in[7]; // [128][64]
  const float* b2  = (const float*)d_in[8];
  float* out = (float*)d_out;

  const int N = GN, E = GE;
  const int FEAT = N * 64;
  float* agg = (float*)d_ws;       // [N*64]
  float* h   = agg + FEAT;         // [N*64]

  const float ninf = -__builtin_huge_valf();
  dim3 blkF(256), grdF((FEAT + 255) / 256);
  dim3 blkE(256), grdE(1024);      // 8192 waves over 100k edge tiles

  for (int g = 0; g < 2; ++g) {
    const float* x  = g ? x2 : x1;
    const int*   ei = g ? ei2 : ei1;
    float* og = out + (size_t)g * FEAT;

    fill_kernel<<<grdF, blkF, 0, stream>>>(agg, FEAT, ninf);
    edgeconv_wmma_kernel<32><<<grdE, blkE, 0, stream>>>(x, ei, W1, b1, agg, E);
    finalize_prelu_kernel<<<grdF, blkF, 0, stream>>>(agg, pa, h, FEAT);

    fill_kernel<<<grdF, blkF, 0, stream>>>(og, FEAT, ninf);
    edgeconv_wmma_kernel<64><<<grdE, blkE, 0, stream>>>(h, ei, W2, b2, og, E);
    finalize_zero_kernel<<<grdF, blkF, 0, stream>>>(og, FEAT);
  }
}
